// hybrid_cnn_59992103190637
// MI455X (gfx1250) — matrix operand order repaired and checked by reading, not compiled, not run on hardware
//
#include <hip/hip_runtime.h>
#include <hip/hip_bf16.h>

typedef __bf16 bf16_t;
typedef __attribute__((ext_vector_type(16))) __bf16 v16bf;
typedef __attribute__((ext_vector_type(8)))  __bf16 v8bf;
typedef __attribute__((ext_vector_type(8)))  float  v8f;

#define B_     16
#define L_     1024
#define D_     512
#define NK_    512
#define H_     8
#define DH_    64
#define NL_    1024
#define AD_    256
#define LC_    1022
#define LP_    1024
#define EROWS_ 1026   // L + 2 taps of zero padding for the K=3 VALID conv

// ---------------------------------------------------------------------------
// WMMA fragment helpers (CDNA5 wave32, v_wmma_f32_16x16x32_bf16)
//
// A (16x32, MxK) per ISA: lane m (0-15): K in [0..7] U [16..23];
//                         lane m+16   : K in [8..15] U [24..31].
// B (32x16, KxN): lane n (0-15): column n, K in [0..7] U [16..23];
//                 lane n+16    : column n, K in [8..15] U [24..31].
//   -> Bt stored as [N][K] row-major gives two 16B loads per lane.
// C/D (16x16 f32): lane group (lane>>4) selects rows 0-7 / 8-15,
//                  col = lane&15, VGPR r = row r (+8).
// ---------------------------------------------------------------------------
static __device__ inline v16bf load_a_frag(const bf16_t* __restrict__ base, int ld, int lane) {
    const int m    = lane & 15;
    const int half = lane >> 4;
    const bf16_t* p = base + (size_t)m * ld + half * 8;
    v8bf lo = *(const v8bf*)(p);
    v8bf hi = *(const v8bf*)(p + 16);
    v16bf a;
#pragma unroll
    for (int i = 0; i < 8; ++i) { a[i] = lo[i]; a[8 + i] = hi[i]; }
    return a;
}

static __device__ inline v16bf load_b_frag(const bf16_t* __restrict__ base, int ld, int lane) {
    const int n    = lane & 15;
    const int half = lane >> 4;
    const bf16_t* p = base + (size_t)n * ld + half * 8;
    v8bf lo = *(const v8bf*)(p);
    v8bf hi = *(const v8bf*)(p + 16);
    v16bf b;
#pragma unroll
    for (int i = 0; i < 8; ++i) { b[i] = lo[i]; b[8 + i] = hi[i]; }
    return b;
}

static __device__ inline v8f wmma_bf16(v16bf a, v16bf b, v8f c) {
    return __builtin_amdgcn_wmma_f32_16x16x32_bf16(false, a, false, b, (short)0, c, false, false);
}

// ---------------------------------------------------------------------------
// Prep kernels: fp32 -> bf16 conversions / layout changes
// ---------------------------------------------------------------------------
__global__ void prep_emb(const float* __restrict__ emb, bf16_t* __restrict__ out) {
    const size_t total = (size_t)B_ * EROWS_ * D_;
    for (size_t idx = (size_t)blockIdx.x * blockDim.x + threadIdx.x;
         idx < total; idx += (size_t)gridDim.x * blockDim.x) {
        size_t d   = idx % D_;
        size_t rb  = idx / D_;
        size_t row = rb % EROWS_;
        size_t b   = rb / EROWS_;
        float v = (row < L_) ? emb[((size_t)b * L_ + row) * D_ + d] : 0.f;
        out[idx] = (bf16_t)v;
    }
}

// w: [NK][D][3]  ->  wt: [3][NK][D] bf16 (each tap is a dense [NK][D] B-matrix)
__global__ void prep_convw(const float* __restrict__ w, bf16_t* __restrict__ wt) {
    const size_t total = (size_t)3 * NK_ * D_;
    for (size_t idx = (size_t)blockIdx.x * blockDim.x + threadIdx.x;
         idx < total; idx += (size_t)gridDim.x * blockDim.x) {
        size_t t   = idx / ((size_t)NK_ * D_);
        size_t rem = idx % ((size_t)NK_ * D_);
        size_t nk  = rem / D_;
        size_t d   = rem % D_;
        wt[idx] = (bf16_t)w[(nk * D_ + d) * 3 + t];
    }
}

__global__ void cvt_bf16(const float* __restrict__ in, bf16_t* __restrict__ out, size_t n) {
    for (size_t idx = (size_t)blockIdx.x * blockDim.x + threadIdx.x;
         idx < n; idx += (size_t)gridDim.x * blockDim.x)
        out[idx] = (bf16_t)in[idx];
}

// ---------------------------------------------------------------------------
// Conv (K=3, VALID) + bias + ELU as 3 accumulated WMMA GEMMs.
// 4-wave blocks: wave w owns a 16(l) x 64(nk) tile; the 4 waves of a block
// cover 64 consecutive l-rows with identical weight (B) streams -> WGP$ reuse.
// K-loop per tap: ping-pong double buffer (no register rotation copies).
// out_lk : [B][LP][NK] bf16 (row-major, l-major) — used for q,k
// out_kl : [B][NK][LP] bf16 (channel-major)       — used for v
// ---------------------------------------------------------------------------
__global__ __launch_bounds__(128) void conv_elu_wmma(
    const bf16_t* __restrict__ emb, const bf16_t* __restrict__ wt,
    const float* __restrict__ bias,
    bf16_t* __restrict__ out_lk, bf16_t* __restrict__ out_kl)
{
    const int lane = threadIdx.x & 31;
    const int wav  = threadIdx.x >> 5;
    const int lt4  = blockIdx.x & 15;       // 16 blocks of 64 l-rows
    const int ns   = blockIdx.x >> 4;       // 8 n strips
    const int b    = blockIdx.z;
    const int l0   = lt4 * 64 + wav * 16;
    const int n0   = ns * 64;
    const bf16_t* embb = emb + (size_t)b * EROWS_ * D_;

    v8f acc[4] = {};
#pragma unroll
    for (int t = 0; t < 3; ++t) {
        const bf16_t* arow = embb + (size_t)(l0 + t) * D_;
        const bf16_t* brow = wt + ((size_t)t * NK_ + n0) * D_;
        v16bf a0, a1, b0[4], b1[4];
        // prologue: k-step 0 into set0
        a0 = load_a_frag(arow, D_, lane);
#pragma unroll
        for (int nt = 0; nt < 4; ++nt)
            b0[nt] = load_b_frag(brow + (size_t)nt * 16 * D_, D_, lane);
        int k0 = 0;
        for (; k0 + 64 < D_; k0 += 64) {
            // load set1 (k0+32), compute set0 (k0)
            a1 = load_a_frag(arow + k0 + 32, D_, lane);
#pragma unroll
            for (int nt = 0; nt < 4; ++nt)
                b1[nt] = load_b_frag(brow + (size_t)nt * 16 * D_ + k0 + 32, D_, lane);
            __builtin_prefetch(arow + k0 + 96, 0, 3);            // global_prefetch_b8 (near)
            __builtin_prefetch(brow + k0 + 96, 0, 3);
#pragma unroll
            for (int nt = 0; nt < 4; ++nt)
                acc[nt] = wmma_bf16(a0, b0[nt], acc[nt]);
            // load set0 (k0+64), compute set1 (k0+32)
            a0 = load_a_frag(arow + k0 + 64, D_, lane);
#pragma unroll
            for (int nt = 0; nt < 4; ++nt)
                b0[nt] = load_b_frag(brow + (size_t)nt * 16 * D_ + k0 + 64, D_, lane);
#pragma unroll
            for (int nt = 0; nt < 4; ++nt)
                acc[nt] = wmma_bf16(a1, b1[nt], acc[nt]);
        }
        // epilogue: steps k0 (set0) and k0+32 (set1)
        a1 = load_a_frag(arow + k0 + 32, D_, lane);
#pragma unroll
        for (int nt = 0; nt < 4; ++nt)
            b1[nt] = load_b_frag(brow + (size_t)nt * 16 * D_ + k0 + 32, D_, lane);
#pragma unroll
        for (int nt = 0; nt < 4; ++nt)
            acc[nt] = wmma_bf16(a0, b0[nt], acc[nt]);
#pragma unroll
        for (int nt = 0; nt < 4; ++nt)
            acc[nt] = wmma_bf16(a1, b1[nt], acc[nt]);
    }
    const int colL = lane & 15, rb = (lane >> 4) * 8;
#pragma unroll
    for (int nt = 0; nt < 4; ++nt) {
        const int col = n0 + nt * 16 + colL;
        const float bv = bias[col];
#pragma unroll
        for (int r = 0; r < 8; ++r) {
            const int row = l0 + rb + r;
            float x = acc[nt][r] + bv;
            float y = x > 0.f ? x : (__expf(x) - 1.f);   // ELU(alpha=1)
            bf16_t v = (bf16_t)y;
            if (out_lk) out_lk[((size_t)b * LP_ + row) * NK_ + col] = v;
            if (out_kl) out_kl[((size_t)b * NK_ + col) * LP_ + row] = v;
        }
    }
}

// ---------------------------------------------------------------------------
// Self-attention per (b,h): one wave owns a 16-row l-tile; score strip 16x1024
// staged in LDS (64KB dynamic -> 4+ blocks per 320KB WGP), exact softmax
// (mask m>=1022), then P @ V^T.  Both WMMA loops ping-pong double buffered.
// ---------------------------------------------------------------------------
__global__ __launch_bounds__(32) void attn_wmma(
    const bf16_t* __restrict__ qlk, const bf16_t* __restrict__ klk,
    const bf16_t* __restrict__ v1,
    bf16_t* __restrict__ cnn1, bf16_t* __restrict__ cnn1T)
{
    extern __shared__ float S[];                 // [16][LP_]
    const int lane = threadIdx.x;
    const int lt = blockIdx.x, h = blockIdx.y, b = blockIdx.z;
    const int l0 = lt * 16;
    const float scale = 0.125f;                  // 1/sqrt(64)

    v16bf aq[2];
#pragma unroll
    for (int kk = 0; kk < 2; ++kk)
        aq[kk] = load_a_frag(qlk + ((size_t)b * LP_ + l0) * NK_ + h * DH_ + kk * 32, NK_, lane);

    const int colL = lane & 15, rb = (lane >> 4) * 8;
    const bf16_t* kbase = klk + (size_t)b * LP_ * NK_ + h * DH_;

    // phase 1: S = (Q^T K) * scale, K-fragments ping-pong over m-tiles
    v16bf bk0[2], bk1[2];
#pragma unroll
    for (int kk = 0; kk < 2; ++kk)
        bk0[kk] = load_b_frag(kbase + kk * 32, NK_, lane);
    int mt = 0;
    for (; mt < 62; mt += 2) {
#pragma unroll
        for (int kk = 0; kk < 2; ++kk)
            bk1[kk] = load_b_frag(kbase + (size_t)(mt + 1) * 16 * NK_ + kk * 32, NK_, lane);
        __builtin_prefetch(kbase + (size_t)(mt + 3) * 16 * NK_, 0, 3);
        {
            v8f c = {};
#pragma unroll
            for (int kk = 0; kk < 2; ++kk) c = wmma_bf16(aq[kk], bk0[kk], c);
#pragma unroll
            for (int r = 0; r < 8; ++r)
                S[(rb + r) * LP_ + mt * 16 + colL] = c[r] * scale;
        }
#pragma unroll
        for (int kk = 0; kk < 2; ++kk)
            bk0[kk] = load_b_frag(kbase + (size_t)(mt + 2) * 16 * NK_ + kk * 32, NK_, lane);
        {
            v8f c = {};
#pragma unroll
            for (int kk = 0; kk < 2; ++kk) c = wmma_bf16(aq[kk], bk1[kk], c);
#pragma unroll
            for (int r = 0; r < 8; ++r)
                S[(rb + r) * LP_ + (mt + 1) * 16 + colL] = c[r] * scale;
        }
    }
    // epilogue: tiles 62 (set0) and 63 (set1)
#pragma unroll
    for (int kk = 0; kk < 2; ++kk)
        bk1[kk] = load_b_frag(kbase + (size_t)63 * 16 * NK_ + kk * 32, NK_, lane);
    {
        v8f c = {};
#pragma unroll
        for (int kk = 0; kk < 2; ++kk) c = wmma_bf16(aq[kk], bk0[kk], c);
#pragma unroll
        for (int r = 0; r < 8; ++r)
            S[(rb + r) * LP_ + 62 * 16 + colL] = c[r] * scale;
    }
    {
        v8f c = {};
#pragma unroll
        for (int kk = 0; kk < 2; ++kk) c = wmma_bf16(aq[kk], bk1[kk], c);
#pragma unroll
        for (int r = 0; r < 8; ++r)
            S[(rb + r) * LP_ + 63 * 16 + colL] = c[r] * scale;
    }
    __syncthreads();

    // phase 2: row softmax (mask padded columns)
    for (int row = 0; row < 16; ++row) {
        float* sr = S + row * LP_;
        float mx = -3.0e38f;
        for (int j = lane; j < LC_; j += 32) mx = fmaxf(mx, sr[j]);
#pragma unroll
        for (int off = 16; off > 0; off >>= 1) mx = fmaxf(mx, __shfl_xor(mx, off, 32));
        float sum = 0.f;
        for (int j = lane; j < LP_; j += 32) {
            float e = (j < LC_) ? __expf(sr[j] - mx) : 0.f;
            sr[j] = e; sum += e;
        }
#pragma unroll
        for (int off = 16; off > 0; off >>= 1) sum += __shfl_xor(sum, off, 32);
        float inv = 1.f / sum;
        for (int j = lane; j < LP_; j += 32) sr[j] *= inv;
    }
    __syncthreads();

    // phase 3: O = P @ V^T, V-fragments ping-pong; P converted from LDS
    const bf16_t* vbase = v1 + ((size_t)b * NK_ + h * DH_) * LP_;
    v8f o[4] = {};
    const int m = lane & 15, half = lane >> 4;
    const float* prow = S + m * LP_ + half * 8;

    v16bf vb0[4], vb1[4];
#pragma unroll
    for (int nt = 0; nt < 4; ++nt)
        vb0[nt] = load_b_frag(vbase + (size_t)nt * 16 * LP_, LP_, lane);
    int k0 = 0;
    for (; k0 + 64 < LP_; k0 += 64) {
#pragma unroll
        for (int nt = 0; nt < 4; ++nt)
            vb1[nt] = load_b_frag(vbase + (size_t)nt * 16 * LP_ + k0 + 32, LP_, lane);
        __builtin_prefetch(vbase + k0 + 96, 0, 3);
        {
            const float* pr = prow + k0;
            v16bf pa;
#pragma unroll
            for (int i = 0; i < 8; ++i) { pa[i] = (bf16_t)pr[i]; pa[8 + i] = (bf16_t)pr[16 + i]; }
#pragma unroll
            for (int nt = 0; nt < 4; ++nt) o[nt] = wmma_bf16(pa, vb0[nt], o[nt]);
        }
#pragma unroll
        for (int nt = 0; nt < 4; ++nt)
            vb0[nt] = load_b_frag(vbase + (size_t)nt * 16 * LP_ + k0 + 64, LP_, lane);
        {
            const float* pr = prow + k0 + 32;
            v16bf pa;
#pragma unroll
            for (int i = 0; i < 8; ++i) { pa[i] = (bf16_t)pr[i]; pa[8 + i] = (bf16_t)pr[16 + i]; }
#pragma unroll
            for (int nt = 0; nt < 4; ++nt) o[nt] = wmma_bf16(pa, vb1[nt], o[nt]);
        }
    }
    // epilogue: steps k0 (set0) and k0+32 (set1)
#pragma unroll
    for (int nt = 0; nt < 4; ++nt)
        vb1[nt] = load_b_frag(vbase + (size_t)nt * 16 * LP_ + k0 + 32, LP_, lane);
    {
        const float* pr = prow + k0;
        v16bf pa;
#pragma unroll
        for (int i = 0; i < 8; ++i) { pa[i] = (bf16_t)pr[i]; pa[8 + i] = (bf16_t)pr[16 + i]; }
#pragma unroll
        for (int nt = 0; nt < 4; ++nt) o[nt] = wmma_bf16(pa, vb0[nt], o[nt]);
    }
    {
        const float* pr = prow + k0 + 32;
        v16bf pa;
#pragma unroll
        for (int i = 0; i < 8; ++i) { pa[i] = (bf16_t)pr[i]; pa[8 + i] = (bf16_t)pr[16 + i]; }
#pragma unroll
        for (int nt = 0; nt < 4; ++nt) o[nt] = wmma_bf16(pa, vb1[nt], o[nt]);
    }

#pragma unroll
    for (int nt = 0; nt < 4; ++nt) {
        const int col = h * DH_ + nt * 16 + colL;
#pragma unroll
        for (int r = 0; r < 8; ++r) {
            const int row = l0 + rb + r;
            bf16_t v = (bf16_t)o[nt][r];
            cnn1 [((size_t)b * LP_ + row) * NK_ + col] = v;
            cnn1T[((size_t)b * NK_ + col) * LP_ + row] = v;
        }
    }
}

// ---------------------------------------------------------------------------
// Generic bf16 WMMA GEMM: block = 4 waves; wave w computes the 16x64 strip at
// rows m0 = blockIdx.x*64 + w*16 (waves share identical B streams -> WGP$).
// K-loop ping-pong double buffered (K must be a multiple of 64; all call
// sites use K in {256, 512, 1024}).
// mode: 0 = f32 row-major, 1 = bf16 row-major, 2 = f32 transposed (C[n][m])
// act : 0 = none, 1 = tanh
// ---------------------------------------------------------------------------
__global__ __launch_bounds__(128) void gemm_bf16_wmma(
    const bf16_t* __restrict__ A, const bf16_t* __restrict__ Bt,
    size_t strideA, size_t strideB,
    float* __restrict__ Cf, bf16_t* __restrict__ Cb, size_t strideC,
    int ldA, int ldB, int ldC, int K, int mode, int act)
{
    const int lane = threadIdx.x & 31;
    const int wav  = threadIdx.x >> 5;
    const int m0 = blockIdx.x * 64 + wav * 16;
    const int n0 = blockIdx.y * 64;
    const int b  = blockIdx.z;
    A  += (size_t)b * strideA;
    Bt += (size_t)b * strideB;
    if (Cf) Cf += (size_t)b * strideC;
    if (Cb) Cb += (size_t)b * strideC;

    const bf16_t* arow = A + (size_t)m0 * ldA;
    const bf16_t* brow = Bt + (size_t)n0 * ldB;
    v8f acc[4] = {};
    v16bf a0, a1, b0[4], b1[4];
    // prologue: k-step 0 into set0
    a0 = load_a_frag(arow, ldA, lane);
#pragma unroll
    for (int nt = 0; nt < 4; ++nt)
        b0[nt] = load_b_frag(brow + (size_t)nt * 16 * ldB, ldB, lane);

    int k0 = 0;
    for (; k0 + 64 < K; k0 += 64) {
        // load set1 (k0+32), compute set0 (k0)
        a1 = load_a_frag(arow + k0 + 32, ldA, lane);
#pragma unroll
        for (int nt = 0; nt < 4; ++nt)
            b1[nt] = load_b_frag(brow + (size_t)nt * 16 * ldB + k0 + 32, ldB, lane);
        __builtin_prefetch(arow + k0 + 96, 0, 3);               // global_prefetch_b8 (near)
        __builtin_prefetch(brow + k0 + 96, 0, 3);
#pragma unroll
        for (int nt = 0; nt < 4; ++nt)
            acc[nt] = wmma_bf16(a0, b0[nt], acc[nt]);
        // load set0 (k0+64), compute set1 (k0+32)
        a0 = load_a_frag(arow + k0 + 64, ldA, lane);
#pragma unroll
        for (int nt = 0; nt < 4; ++nt)
            b0[nt] = load_b_frag(brow + (size_t)nt * 16 * ldB + k0 + 64, ldB, lane);
#pragma unroll
        for (int nt = 0; nt < 4; ++nt)
            acc[nt] = wmma_bf16(a1, b1[nt], acc[nt]);
    }
    // epilogue: steps k0 (set0) and k0+32 (set1)
    a1 = load_a_frag(arow + k0 + 32, ldA, lane);
#pragma unroll
    for (int nt = 0; nt < 4; ++nt)
        b1[nt] = load_b_frag(brow + (size_t)nt * 16 * ldB + k0 + 32, ldB, lane);
#pragma unroll
    for (int nt = 0; nt < 4; ++nt)
        acc[nt] = wmma_bf16(a0, b0[nt], acc[nt]);
#pragma unroll
    for (int nt = 0; nt < 4; ++nt)
        acc[nt] = wmma_bf16(a1, b1[nt], acc[nt]);

    const int colL = lane & 15, rb = (lane >> 4) * 8;
#pragma unroll
    for (int nt = 0; nt < 4; ++nt) {
        const int col = n0 + nt * 16 + colL;
#pragma unroll
        for (int r = 0; r < 8; ++r) {
            const int row = m0 + rb + r;
            float x = acc[nt][r];
            if (act == 1) x = tanhf(x);
            if (mode == 0)      Cf[(size_t)row * ldC + col] = x;
            else if (mode == 1) Cb[(size_t)row * ldC + col] = (bf16_t)x;
            else                Cf[(size_t)col * ldC + row] = x;
        }
    }
}

// ---------------------------------------------------------------------------
// Row softmax over contiguous rows of `width`, masking j >= valid; bf16 out.
// 8 waves per block, one wave per row.
// ---------------------------------------------------------------------------
__global__ __launch_bounds__(256) void softmax_rows(
    const float* __restrict__ in, bf16_t* __restrict__ out, int width, int valid)
{
    const int wave = threadIdx.x >> 5, lane = threadIdx.x & 31;
    const size_t row = (size_t)blockIdx.x * 8 + wave;
    const float* sr = in + row * width;
    bf16_t* orow = out + row * width;
    float mx = -3.0e38f;
    for (int j = lane; j < valid; j += 32) mx = fmaxf(mx, sr[j]);
#pragma unroll
    for (int off = 16; off > 0; off >>= 1) mx = fmaxf(mx, __shfl_xor(mx, off, 32));
    float sum = 0.f;
    for (int j = lane; j < valid; j += 32) sum += __expf(sr[j] - mx);
#pragma unroll
    for (int off = 16; off > 0; off >>= 1) sum += __shfl_xor(sum, off, 32);
    const float inv = 1.f / sum;
    for (int j = lane; j < width; j += 32)
        orow[j] = (bf16_t)((j < valid) ? __expf(sr[j] - mx) * inv : 0.f);
}

// ---------------------------------------------------------------------------
// In-place L2 normalization of the two 512-wide halves of each 1024-wide row.
// ---------------------------------------------------------------------------
__global__ __launch_bounds__(256) void l2norm_halves(float* __restrict__ out)
{
    const int wave = threadIdx.x >> 5, lane = threadIdx.x & 31;
    const size_t row = (size_t)blockIdx.x * 8 + wave;
    float* pr = out + row * (2 * NK_);
#pragma unroll
    for (int hlf = 0; hlf < 2; ++hlf) {
        float* ph = pr + hlf * NK_;
        float s = 0.f;
        for (int j = lane; j < NK_; j += 32) { float x = ph[j]; s += x * x; }
#pragma unroll
        for (int off = 16; off > 0; off >>= 1) s += __shfl_xor(s, off, 32);
        const float inv = 1.f / fmaxf(sqrtf(s), 1e-12f);
        for (int j = lane; j < NK_; j += 32) ph[j] *= inv;
    }
}

// ---------------------------------------------------------------------------
// Host-side orchestration
// ---------------------------------------------------------------------------
extern "C" void kernel_launch(void* const* d_in, const int* in_sizes, int n_in,
                              void* d_out, int out_size, void* d_ws, size_t ws_size,
                              hipStream_t stream) {
    (void)in_sizes; (void)n_in; (void)out_size; (void)ws_size;
    const float* embedding = (const float*)d_in[0];
    const float* label_emb = (const float*)d_in[1];
    const float* wq        = (const float*)d_in[2];
    const float* bq        = (const float*)d_in[3];
    const float* wk        = (const float*)d_in[4];
    const float* bk        = (const float*)d_in[5];
    const float* wv        = (const float*)d_in[6];
    const float* bv        = (const float*)d_in[7];
    const float* attn_w1   = (const float*)d_in[8];
    const float* attn_w2   = (const float*)d_in[9];
    float* dout = (float*)d_out;

    // workspace carve-out (all regions fully rewritten every call)
    char* p = (char*)d_ws;
    auto alloc = [&](size_t bytes) -> char* {
        char* r = p; p += (bytes + 255) & ~(size_t)255; return r;
    };
    bf16_t* emb_bf  = (bf16_t*)alloc((size_t)B_ * EROWS_ * D_ * 2);
    bf16_t* wqt     = (bf16_t*)alloc((size_t)3 * NK_ * D_ * 2);
    bf16_t* wkt     = (bf16_t*)alloc((size_t)3 * NK_ * D_ * 2);
    bf16_t* wvt     = (bf16_t*)alloc((size_t)3 * NK_ * D_ * 2);
    bf16_t* lab_bf  = (bf16_t*)alloc((size_t)NL_ * NK_ * 2);
    bf16_t* w1_bf   = (bf16_t*)alloc((size_t)AD_ * NK_ * 2);
    bf16_t* w2_bf   = (bf16_t*)alloc((size_t)NL_ * AD_ * 2);
    bf16_t* qlk     = (bf16_t*)alloc((size_t)B_ * LP_ * NK_ * 2);
    bf16_t* klk     = (bf16_t*)alloc((size_t)B_ * LP_ * NK_ * 2);
    bf16_t* v1      = (bf16_t*)alloc((size_t)B_ * NK_ * LP_ * 2);
    bf16_t* cnn1    = (bf16_t*)alloc((size_t)B_ * LP_ * NK_ * 2);
    bf16_t* cnn1T   = (bf16_t*)alloc((size_t)B_ * NK_ * LP_ * 2);
    bf16_t* T1      = (bf16_t*)alloc((size_t)B_ * LP_ * AD_ * 2);
    float*  scoreT  = (float*) alloc((size_t)B_ * NL_ * LP_ * 4);  // reused for b2
    bf16_t* P1      = (bf16_t*)alloc((size_t)B_ * NL_ * LP_ * 2);
    bf16_t* P2      = (bf16_t*)alloc((size_t)B_ * NL_ * LP_ * 2);

    // ---- prep: bf16 conversions ----
    {
        size_t n = (size_t)B_ * EROWS_ * D_;
        prep_emb<<<dim3((unsigned)((n + 255) / 256)), 256, 0, stream>>>(embedding, emb_bf);
        size_t nw = (size_t)3 * NK_ * D_;
        prep_convw<<<dim3((unsigned)((nw + 255) / 256)), 256, 0, stream>>>(wq, wqt);
        prep_convw<<<dim3((unsigned)((nw + 255) / 256)), 256, 0, stream>>>(wk, wkt);
        prep_convw<<<dim3((unsigned)((nw + 255) / 256)), 256, 0, stream>>>(wv, wvt);
        size_t nl = (size_t)NL_ * NK_;
        cvt_bf16<<<dim3((unsigned)((nl + 255) / 256)), 256, 0, stream>>>(label_emb, lab_bf, nl);
        size_t n1 = (size_t)AD_ * NK_;
        cvt_bf16<<<dim3((unsigned)((n1 + 255) / 256)), 256, 0, stream>>>(attn_w1, w1_bf, n1);
        size_t n2 = (size_t)NL_ * AD_;
        cvt_bf16<<<dim3((unsigned)((n2 + 255) / 256)), 256, 0, stream>>>(attn_w2, w2_bf, n2);
    }

    // ---- conv + ELU (q -> qlk, k -> klk, v -> v1 channel-major) ----
    {
        dim3 g(128, 1, B_);   // 16 l-blocks (64 rows) x 8 n-strips
        conv_elu_wmma<<<g, 128, 0, stream>>>(emb_bf, wqt, bq, qlk, (bf16_t*)nullptr);
        conv_elu_wmma<<<g, 128, 0, stream>>>(emb_bf, wkt, bk, klk, (bf16_t*)nullptr);
        conv_elu_wmma<<<g, 128, 0, stream>>>(emb_bf, wvt, bv, (bf16_t*)nullptr, v1);
    }

    // ---- multi-head self-attention -> cnn1 / cnn1T ----
    attn_wmma<<<dim3(64, H_, B_), 32, 16 * LP_ * sizeof(float), stream>>>(qlk, klk, v1, cnn1, cnn1T);

    // ---- label attention: T1 = tanh(cnn1 @ W1^T) ----
    gemm_bf16_wmma<<<dim3(LP_ / 64, AD_ / 64, B_), 128, 0, stream>>>(
        cnn1, w1_bf, (size_t)LP_ * NK_, 0,
        (float*)nullptr, T1, (size_t)LP_ * AD_,
        NK_, NK_, AD_, NK_, /*mode=*/1, /*act=*/1);

    // scoreT[b][n][l] = (T1 @ W2^T)^T  (transposed store for contiguous softmax)
    gemm_bf16_wmma<<<dim3(LP_ / 64, NL_ / 64, B_), 128, 0, stream>>>(
        T1, w2_bf, (size_t)LP_ * AD_, 0,
        scoreT, (bf16_t*)nullptr, (size_t)NL_ * LP_,
        AD_, AD_, LP_, AD_, /*mode=*/2, /*act=*/0);

    // P1 = softmax over l (mask l>=1022)
    softmax_rows<<<dim3((B_ * NL_) / 8), 256, 0, stream>>>(scoreT, P1, LP_, LC_);

    // ---- cross-attention scores: b2[b][n][l] = label_emb @ k1 (reuses scoreT) ----
    gemm_bf16_wmma<<<dim3(NL_ / 64, LP_ / 64, B_), 128, 0, stream>>>(
        lab_bf, klk, 0, (size_t)LP_ * NK_,
        scoreT, (bf16_t*)nullptr, (size_t)NL_ * LP_,
        NK_, NK_, LP_, NK_, /*mode=*/0, /*act=*/0);

    softmax_rows<<<dim3((B_ * NL_) / 8), 256, 0, stream>>>(scoreT, P2, LP_, LC_);

    // ---- out1 = P1 @ cnn1 -> d_out[:, :, 0:512] ----
    gemm_bf16_wmma<<<dim3(NL_ / 64, NK_ / 64, B_), 128, 0, stream>>>(
        P1, cnn1T, (size_t)NL_ * LP_, (size_t)NK_ * LP_,
        dout, (bf16_t*)nullptr, (size_t)NL_ * 2 * NK_,
        LP_, LP_, 2 * NK_, LP_, /*mode=*/0, /*act=*/0);

    // ---- out2 = P2 @ v1^T -> d_out[:, :, 512:1024] ----
    gemm_bf16_wmma<<<dim3(NL_ / 64, NK_ / 64, B_), 128, 0, stream>>>(
        P2, v1, (size_t)NL_ * LP_, (size_t)NK_ * LP_,
        dout + NK_, (bf16_t*)nullptr, (size_t)NL_ * 2 * NK_,
        LP_, LP_, 2 * NK_, LP_, /*mode=*/0, /*act=*/0);

    // ---- in-place L2 norm of both halves ----
    l2norm_halves<<<dim3((B_ * NL_) / 8), 256, 0, stream>>>(dout);
}
